// MultiHeadAttention_73632919323240
// MI455X (gfx1250) — matrix operand order repaired and checked by reading, not compiled, not run on hardware
//
#include <hip/hip_runtime.h>

// ---------------------------------------------------------------------------
// MI455X (gfx1250) sigmoid multi-head attention, bf16 WMMA everywhere.
// Shapes: B=2, S=2048, D=1024, H=16, dk=64.
// d_out = [H_out (B*S*D f32)] ++ [A (B*H*S*S f32)]
// d_ws  = Q,K,V,Hcat as bf16 (4 x B*S*D elements)
// Data movement: async global->LDS (ASYNCcnt) for raw bf16 tile staging,
// global_prefetch for next tiles, LDS b128 fragment loads feeding WMMA.
// ---------------------------------------------------------------------------

typedef __attribute__((ext_vector_type(16))) __bf16 bf16x16;
typedef __attribute__((ext_vector_type(8)))  __bf16 bf16x8;
typedef __attribute__((ext_vector_type(8)))  float  floatx8;

static constexpr int BATCH  = 2;
static constexpr int SEQ    = 2048;
static constexpr int DMODEL = 1024;
static constexpr int HEADS  = 16;
static constexpr int DK     = 64;
static constexpr int MROWS  = BATCH * SEQ;          // 4096 GEMM rows

#define WMMA_BF16(a, b, c) \
  __builtin_amdgcn_wmma_f32_16x16x32_bf16(false, (a), false, (b), (short)0, (c), false, false)

// Async copy of 16 bytes global -> LDS, per lane. Tracked by ASYNCcnt.
__device__ __forceinline__ void async_copy_b128(const void* gsrc, void* ldst) {
  unsigned long long ga = (unsigned long long)(uintptr_t)gsrc;
  unsigned la = (unsigned)(uintptr_t)ldst;        // low 32 bits = LDS byte offset
  asm volatile("global_load_async_to_lds_b128 %0, %1, off"
               :: "v"(la), "v"(ga) : "memory");
}
__device__ __forceinline__ void async_wait0() {
  asm volatile("s_wait_asynccnt 0" ::: "memory");
}

// ---- WMMA fragment loaders (16x16x32 bf16, wave32 layouts per ISA 7.12.2) --
// A 16x32: lane<16 -> row=lane, K chunks {kh*8..+8} and {16+kh*8..+8}
__device__ __forceinline__ bf16x16 load_frag_a(const __bf16* base, int ld) {
  const int lane = threadIdx.x & 31;
  const int row  = lane & 15;
  const int kh   = lane >> 4;
  const __bf16* p = base + row * ld + kh * 8;
  bf16x8 c0 = *(const bf16x8*)(p);
  bf16x8 c1 = *(const bf16x8*)(p + 16);
  return __builtin_shufflevector(c0, c1, 0, 1, 2, 3, 4, 5, 6, 7,
                                         8, 9, 10, 11, 12, 13, 14, 15);
}

// B 32x16 from a transposed (N-major) buffer: element (k,n) at baseT[n*ld+k].
__device__ __forceinline__ bf16x16 load_frag_b(const __bf16* baseT, int ld) {
  const int lane = threadIdx.x & 31;
  const int col  = lane & 15;
  const int kh   = lane >> 4;
  const __bf16* p = baseT + col * ld + kh * 8;
  bf16x8 c0 = *(const bf16x8*)(p);
  bf16x8 c1 = *(const bf16x8*)(p + 16);
  return __builtin_shufflevector(c0, c1, 0, 1, 2, 3, 4, 5, 6, 7,
                                         8, 9, 10, 11, 12, 13, 14, 15);
}

// ---------------------------------------------------------------------------
// GEMM 1: C_bf16[M,N] = alpha * (A_f32[M,K] @ B_f32[K,N])
// 128x128 tile per 256-thread block (8 waves, each 64x32), BK=64.
// ---------------------------------------------------------------------------
__global__ __launch_bounds__(256) void gemm_f32_bf16(
    const float* __restrict__ A, const float* __restrict__ Bm,
    __bf16* __restrict__ C, int M, int N, int K, float alpha) {
  constexpr int BM = 128, BN = 128, BK = 64;
  __shared__ __align__(16) __bf16 As[BM * BK];   // [m][k]
  __shared__ __align__(16) __bf16 Bt[BN * BK];   // [n][k]  (transposed)
  const int m0 = blockIdx.y * BM, n0 = blockIdx.x * BN;
  const int t = threadIdx.x, w = t >> 5;
  const int wm = w & 1, wn = w >> 1;             // wave -> 64x32 sub-tile

  floatx8 acc[4][2] = {};
  for (int k0 = 0; k0 < K; k0 += BK) {
    { // stage A: 128x64, 32 elems/thread, fp32 -> bf16
      const int r = t >> 1, c = (t & 1) * 32;
      const float* src = A + (size_t)(m0 + r) * K + k0 + c;
      __bf16* dst = As + r * BK + c;
#pragma unroll
      for (int i = 0; i < 32; ++i) dst[i] = (__bf16)src[i];
    }
    { // stage B transposed: 64x128, 32 elems/thread
      const int kk = t >> 2, nn = (t & 3) * 32;
      const float* src = Bm + (size_t)(k0 + kk) * N + n0 + nn;
#pragma unroll
      for (int i = 0; i < 32; ++i) Bt[(nn + i) * BK + kk] = (__bf16)src[i];
    }
    __syncthreads();
#pragma unroll
    for (int ks = 0; ks < 2; ++ks) {
      bf16x16 af[4], bfv[2];
#pragma unroll
      for (int i = 0; i < 4; ++i)
        af[i] = load_frag_a(As + (wm * 64 + i * 16) * BK + ks * 32, BK);
#pragma unroll
      for (int j = 0; j < 2; ++j)
        bfv[j] = load_frag_b(Bt + (wn * 32 + j * 16) * BK + ks * 32, BK);
#pragma unroll
      for (int i = 0; i < 4; ++i)
#pragma unroll
        for (int j = 0; j < 2; ++j)
          acc[i][j] = WMMA_BF16(af[i], bfv[j], acc[i][j]);
    }
    __syncthreads();
  }
  const int lane = t & 31;
  const int nin = lane & 15, mofs = (lane >> 4) * 8;
#pragma unroll
  for (int i = 0; i < 4; ++i)
#pragma unroll
    for (int j = 0; j < 2; ++j)
#pragma unroll
      for (int v = 0; v < 8; ++v) {
        const int row = m0 + wm * 64 + i * 16 + mofs + v;
        const int col = n0 + wn * 32 + j * 16 + nin;
        C[(size_t)row * N + col] = (__bf16)(acc[i][j][v] * alpha);
      }
}

// ---------------------------------------------------------------------------
// GEMM 2: C_f32[M,N] = A_bf16[M,K] @ B_f32[K,N] + bias[N]
// A tile staged with async global->LDS copies (raw bf16 bytes).
// ---------------------------------------------------------------------------
__global__ __launch_bounds__(256) void gemm_bf16_f32_bias(
    const __bf16* __restrict__ A, const float* __restrict__ Bm,
    const float* __restrict__ bias, float* __restrict__ C,
    int M, int N, int K) {
  constexpr int BM = 128, BN = 128, BK = 64;
  __shared__ __align__(16) __bf16 As[BM * BK];
  __shared__ __align__(16) __bf16 Bt[BN * BK];
  const int m0 = blockIdx.y * BM, n0 = blockIdx.x * BN;
  const int t = threadIdx.x, w = t >> 5;
  const int wm = w & 1, wn = w >> 1;

  floatx8 acc[4][2] = {};
  for (int k0 = 0; k0 < K; k0 += BK) {
    { // stage A via async DMA: 4x 16B per thread
      const int r = t >> 1, c = (t & 1) * 32;
      const __bf16* src = A + (size_t)(m0 + r) * K + k0 + c;
      __bf16* dst = As + r * BK + c;
#pragma unroll
      for (int i = 0; i < 4; ++i) async_copy_b128(src + i * 8, dst + i * 8);
    }
    { // stage B transposed with fp32->bf16 convert (manual)
      const int kk = t >> 2, nn = (t & 3) * 32;
      const float* src = Bm + (size_t)(k0 + kk) * N + n0 + nn;
#pragma unroll
      for (int i = 0; i < 32; ++i) Bt[(nn + i) * BK + kk] = (__bf16)src[i];
    }
    async_wait0();
    __syncthreads();
#pragma unroll
    for (int ks = 0; ks < 2; ++ks) {
      bf16x16 af[4], bfv[2];
#pragma unroll
      for (int i = 0; i < 4; ++i)
        af[i] = load_frag_a(As + (wm * 64 + i * 16) * BK + ks * 32, BK);
#pragma unroll
      for (int j = 0; j < 2; ++j)
        bfv[j] = load_frag_b(Bt + (wn * 32 + j * 16) * BK + ks * 32, BK);
#pragma unroll
      for (int i = 0; i < 4; ++i)
#pragma unroll
        for (int j = 0; j < 2; ++j)
          acc[i][j] = WMMA_BF16(af[i], bfv[j], acc[i][j]);
    }
    __syncthreads();
  }
  const int lane = t & 31;
  const int nin = lane & 15, mofs = (lane >> 4) * 8;
#pragma unroll
  for (int i = 0; i < 4; ++i)
#pragma unroll
    for (int j = 0; j < 2; ++j) {
      const int col = n0 + wn * 32 + j * 16 + nin;
      const float bb = bias[col];
#pragma unroll
      for (int v = 0; v < 8; ++v) {
        const int row = m0 + wm * 64 + i * 16 + mofs + v;
        C[(size_t)row * N + col] = acc[i][j][v] + bb;
      }
    }
}

// ---------------------------------------------------------------------------
// Fused sigmoid attention per (b, h, 64-row q-tile):
//   scores = (Q/sqrt(dk)) @ K^T  (scale folded into Q already)
//   A = sigmoid(scores)  -> streamed fp32 to d_out, bf16 to LDS
//   H += A @ V           -> bf16 to Hws in [B,S,D] layout
// Q fragments live in registers for the entire k loop. K tiles staged with
// async global->LDS DMA; next K/V tiles prefetched while WMMA runs.
// Score waves own 32x32 sub-tiles so each A-output row gets 128B-contiguous
// wave stores (A is the 537MB dominant traffic stream).
// ---------------------------------------------------------------------------
__global__ __launch_bounds__(256) void attn_sigmoid_kernel(
    const __bf16* __restrict__ Q, const __bf16* __restrict__ K,
    const __bf16* __restrict__ V, float* __restrict__ Aout,
    __bf16* __restrict__ Hws) {
  constexpr int QT = 64, KT = 128;
  __shared__ __align__(16) __bf16 Ks[KT * DK];   // [kpos][d]   (= B^T for scores)
  __shared__ __align__(16) __bf16 Vt[DK * KT];   // [d][kpos]   (= B^T for A@V)
  __shared__ __align__(16) __bf16 As[QT * KT];   // [q][kpos]   sigmoid scores bf16

  const int bh = blockIdx.y;
  const int b = bh >> 4, h = bh & 15;
  const int q0 = blockIdx.x * QT;
  const int t = threadIdx.x, w = t >> 5, lane = t & 31;

  const __bf16* Qbase = Q + ((size_t)b * SEQ + q0) * DMODEL + h * DK;
  const __bf16* Kbase = K + (size_t)b * SEQ * DMODEL + h * DK;
  const __bf16* Vbase = V + (size_t)b * SEQ * DMODEL + h * DK;
  float* Abase = Aout + ((size_t)bh * SEQ + q0) * SEQ;

  // Score-phase wave mapping: 32x32 sub-tile.
  const int wi = w >> 2;                 // row group (2 x 32 rows)
  const int wj = w & 3;                  // col group (4 x 32 cols)

  // Hoist Q fragments: 2 k-steps x 2 row sub-tiles, reused for all 16 k-tiles.
  bf16x16 qf[2][2];
#pragma unroll
  for (int ks = 0; ks < 2; ++ks)
#pragma unroll
    for (int i = 0; i < 2; ++i)
      qf[ks][i] = load_frag_a(Qbase + (size_t)(wi * 32 + i * 16) * DMODEL + ks * 32, DMODEL);

  // H-phase wave mapping: rows (w&3)*16, cols (w>>2)*32 (2 sub-tiles)
  const int wm3 = w & 3, wn3 = w >> 2;
  floatx8 hacc[2] = {};

  const int nin = lane & 15, mofs = (lane >> 4) * 8;
  const int stg_r = t >> 1, stg_half = t & 1;    // staging assignment

  for (int kt = 0; kt < SEQ / KT; ++kt) {
    { // stage K tile row-major via async DMA (already B^T layout for scores)
      const __bf16* src = Kbase + (size_t)(kt * KT + stg_r) * DMODEL + stg_half * 32;
      __bf16* dst = Ks + stg_r * DK + stg_half * 32;
#pragma unroll
      for (int i = 0; i < 4; ++i) async_copy_b128(src + i * 8, dst + i * 8);
    }
    { // stage V tile transposed (manual scatter)
      const __bf16* src = Vbase + (size_t)(kt * KT + stg_r) * DMODEL + stg_half * 32;
#pragma unroll
      for (int i = 0; i < 32; ++i) Vt[(stg_half * 32 + i) * KT + stg_r] = src[i];
    }
    if (kt + 1 < SEQ / KT) { // prefetch next tiles into cache while WMMA runs
      const size_t nrow = (size_t)((kt + 1) * KT + stg_r) * DMODEL + stg_half * 32;
      __builtin_prefetch(Kbase + nrow, 0, 3);
      __builtin_prefetch(Vbase + nrow, 0, 3);
    }
    async_wait0();
    __syncthreads();

    // ---- scores: wave (wi,wj) owns rows wi*32..+32, cols wj*32..+32 ----
    floatx8 sc[2][2] = {};
#pragma unroll
    for (int ks = 0; ks < 2; ++ks) {
      bf16x16 bk[2];
#pragma unroll
      for (int j = 0; j < 2; ++j)
        bk[j] = load_frag_b(Ks + (wj * 32 + j * 16) * DK + ks * 32, DK);
#pragma unroll
      for (int i = 0; i < 2; ++i)
#pragma unroll
        for (int j = 0; j < 2; ++j)
          sc[i][j] = WMMA_BF16(qf[ks][i], bk[j], sc[i][j]);
    }

    // ---- sigmoid, stream A to HBM (128B-contiguous per row), LDS copy ----
#pragma unroll
    for (int i = 0; i < 2; ++i)
#pragma unroll
      for (int j = 0; j < 2; ++j)
#pragma unroll
        for (int v = 0; v < 8; ++v) {
          const int ml = wi * 32 + i * 16 + mofs + v;
          const int nloc = wj * 32 + j * 16 + nin;
          const float s = 1.0f / (1.0f + __expf(-sc[i][j][v]));
          Abase[(size_t)ml * SEQ + (size_t)kt * KT + nloc] = s;
          As[ml * KT + nloc] = (__bf16)s;
        }
    __syncthreads();

    // ---- H += A @ V over this 128-wide k-tile (4 k-steps) ----
#pragma unroll
    for (int ks2 = 0; ks2 < 4; ++ks2) {
      bf16x16 aa = load_frag_a(As + (wm3 * 16) * KT + ks2 * 32, KT);
#pragma unroll
      for (int j = 0; j < 2; ++j) {
        bf16x16 bv = load_frag_b(Vt + (wn3 * 32 + j * 16) * KT + ks2 * 32, KT);
        hacc[j] = WMMA_BF16(aa, bv, hacc[j]);
      }
    }
    __syncthreads();
  }

  // store H tile -> Hws in [B,S,D] bf16 (direct A-operand layout for GEMM 2)
#pragma unroll
  for (int j = 0; j < 2; ++j)
#pragma unroll
    for (int v = 0; v < 8; ++v) {
      const int ml = wm3 * 16 + mofs + v;
      const int col = wn3 * 32 + j * 16 + nin;
      Hws[((size_t)b * SEQ + q0 + ml) * DMODEL + h * DK + col] = (__bf16)hacc[j][v];
    }
}

// ---------------------------------------------------------------------------
extern "C" void kernel_launch(void* const* d_in, const int* in_sizes, int n_in,
                              void* d_out, int out_size, void* d_ws, size_t ws_size,
                              hipStream_t stream) {
  (void)in_sizes; (void)n_in; (void)out_size; (void)ws_size;
  const float* X_q  = (const float*)d_in[0];
  const float* X_k  = (const float*)d_in[1];
  const float* X_v  = (const float*)d_in[2];
  const float* W_q  = (const float*)d_in[3];
  const float* W_k  = (const float*)d_in[4];
  const float* W_v  = (const float*)d_in[5];
  const float* W_h  = (const float*)d_in[6];
  const float* b_h  = (const float*)d_in[7];

  float* H_out = (float*)d_out;
  float* A_out = (float*)d_out + (size_t)MROWS * DMODEL;       // after H_out

  const size_t NE = (size_t)MROWS * DMODEL;                    // 4,194,304
  __bf16* Qws = (__bf16*)d_ws;
  __bf16* Kws = Qws + NE;
  __bf16* Vws = Kws + NE;
  __bf16* Hws = Vws + NE;

  const dim3 blk(256);
  const dim3 ggrid(DMODEL / 128, MROWS / 128);                 // (8, 32)
  const float qscale = 0.125f;                                 // 1/sqrt(dk)

  gemm_f32_bf16<<<ggrid, blk, 0, stream>>>(X_q, W_q, Qws, MROWS, DMODEL, DMODEL, qscale);
  gemm_f32_bf16<<<ggrid, blk, 0, stream>>>(X_k, W_k, Kws, MROWS, DMODEL, DMODEL, 1.0f);
  gemm_f32_bf16<<<ggrid, blk, 0, stream>>>(X_v, W_v, Vws, MROWS, DMODEL, DMODEL, 1.0f);

  const dim3 agrid(SEQ / 64, BATCH * HEADS);                   // (32, 32)
  attn_sigmoid_kernel<<<agrid, blk, 0, stream>>>(Qws, Kws, Vws, A_out, Hws);

  gemm_bf16_f32_bias<<<ggrid, blk, 0, stream>>>(Hws, W_h, b_h, H_out, MROWS, DMODEL, DMODEL);
}
